// LSTMCustomStateful_70411693851114
// MI455X (gfx1250) — matrix operand order repaired and checked by reading, not compiled, not run on hardware
//
#include <hip/hip_runtime.h>

typedef __attribute__((ext_vector_type(8)))  __bf16 v8bf;
typedef __attribute__((ext_vector_type(16))) __bf16 v16bf;
typedef __attribute__((ext_vector_type(8)))  float  v8f;

// Problem dims
constexpr int BB = 128, SS = 512, II = 256, HH = 1024, OO = 256;
constexpr int KC = II + HH;              // 1280 concat K
constexpr int M1 = 16, M2 = 8;

// K chunking for the LDS-staged recurrent GEMM
constexpr int KCH  = 256;                // chunk 0 = x part, chunks 1..4 = h part
constexpr int NCHK = KC / KCH;           // 5
constexpr int KCHP = KCH + 8;            // padded LDS row stride (528B -> bank stride 4)
constexpr int WROWS = 3 * 16;            // 3 gates x 16 columns per block

// Workspace layout (bytes, all 256-aligned)
constexpr size_t N_WTG  = 3ull * HH * KC;             // gate weights [3][1024][1280] bf16
constexpr size_t N_WT1  = (size_t)M1 * KC;            // W1^T         [16][1280]      bf16
constexpr size_t N_WOUT = (size_t)OO * HH;            // W_out^T      [256][1024]     bf16
constexpr size_t N_XBF  = (size_t)SS * BB * II;       // x bf16       [t][b][k]
constexpr size_t N_HS   = (size_t)(SS + 1) * BB * HH; // h history    [t+1][b][n] bf16
constexpr size_t N_C    = (size_t)BB * HH;            // c state f32
constexpr size_t N_A2   = (size_t)BB * M2;            // a2 f32

constexpr size_t OFF_WTG  = 0;
constexpr size_t OFF_WT1  = OFF_WTG  + N_WTG  * 2;
constexpr size_t OFF_WOUT = OFF_WT1  + N_WT1  * 2;
constexpr size_t OFF_XBF  = OFF_WOUT + N_WOUT * 2;
constexpr size_t OFF_HS   = OFF_XBF  + N_XBF  * 2;
constexpr size_t OFF_C    = OFF_HS   + N_HS   * 2;
constexpr size_t OFF_A2   = OFF_C    + N_C    * 4;

// ---------------- WMMA fragment helpers (contiguous-k 16-bit layouts) ----------------
// A 16x32 bf16: lane m = lane&15; K = k0 + 16*(lane>>4)..+15 (contiguous, matches B)
__device__ __forceinline__ v16bf load_a16(const __bf16* rowbase, int k0, int hi) {
  const v8bf lo = *(const v8bf*)(rowbase + k0 + hi * 16);
  const v8bf hh = *(const v8bf*)(rowbase + k0 + hi * 16 + 8);
  v16bf a;
#pragma unroll
  for (int i = 0; i < 8; ++i) { a[i] = lo[i]; a[i + 8] = hh[i]; }
  return a;
}
// B 32x16 bf16 from N-major storage: lane n = lane&15, K = k0 + 16*(lane>>4)..+15.
// Two 16B loads (works for 16B-aligned padded LDS rows too).
__device__ __forceinline__ v16bf load_b16(const __bf16* p) {
  const v8bf lo = *(const v8bf*)(p);
  const v8bf hh = *(const v8bf*)(p + 8);
  v16bf b;
#pragma unroll
  for (int i = 0; i < 8; ++i) { b[i] = lo[i]; b[i + 8] = hh[i]; }
  return b;
}
__device__ __forceinline__ v8f wmma_bf16(v16bf a, v16bf b, v8f c) {
  return __builtin_amdgcn_wmma_f32_16x16x32_bf16(false, a, false, b, (short)0, c, false, false);
}
__device__ __forceinline__ float sigmoidf_(float x) { return 1.0f / (1.0f + __expf(-x)); }

// CDNA5 async copy: global -> LDS, 16 bytes per lane, tracked by ASYNCcnt.
__device__ __forceinline__ void async_g2l_b128(uint32_t lds_off, const __bf16* gaddr) {
  asm volatile("global_load_async_to_lds_b128 %0, %1, off" :: "v"(lds_off), "v"(gaddr) : "memory");
}
__device__ __forceinline__ void wait_async_le1() {
  asm volatile("s_wait_asynccnt 0x1" ::: "memory");
}
__device__ __forceinline__ void wait_async_0() {
  asm volatile("s_wait_asynccnt 0x0" ::: "memory");
}

// ---------------- prep kernels ----------------
__global__ void k_pack_gate_weights(const float* __restrict__ Wxi, const float* __restrict__ Whi,
                                    const float* __restrict__ Wxc, const float* __restrict__ Whc,
                                    const float* __restrict__ Wxo, const float* __restrict__ Who,
                                    __bf16* __restrict__ wtg) {
  size_t idx = (size_t)blockIdx.x * blockDim.x + threadIdx.x;
  if (idx >= N_WTG) return;
  int g = (int)(idx / ((size_t)HH * KC));
  size_t r = idx % ((size_t)HH * KC);
  int n = (int)(r / KC), k = (int)(r % KC);
  const float* Wx = (g == 0) ? Wxi : (g == 1) ? Wxc : Wxo;
  const float* Wh = (g == 0) ? Whi : (g == 1) ? Whc : Who;
  float v = (k < II) ? Wx[(size_t)k * HH + n] : Wh[(size_t)(k - II) * HH + n];
  wtg[idx] = (__bf16)v;   // layout [g][n][k]
}

__global__ void k_pack_w1(const float* __restrict__ W1, __bf16* __restrict__ wt1) {
  size_t idx = (size_t)blockIdx.x * blockDim.x + threadIdx.x;
  if (idx >= N_WT1) return;
  int n = (int)(idx / KC), k = (int)(idx % KC);
  wt1[idx] = (__bf16)W1[(size_t)k * M1 + n];
}

__global__ void k_pack_wout(const float* __restrict__ Wout, __bf16* __restrict__ woutT) {
  size_t idx = (size_t)blockIdx.x * blockDim.x + threadIdx.x;
  if (idx >= N_WOUT) return;
  int n = (int)(idx / HH), k = (int)(idx % HH);
  woutT[idx] = (__bf16)Wout[(size_t)k * OO + n];
}

__global__ void k_pack_x(const float* __restrict__ seq, __bf16* __restrict__ xbf) {
  size_t idx = (size_t)blockIdx.x * blockDim.x + threadIdx.x;
  if (idx >= N_XBF) return;
  int t = (int)(idx / ((size_t)BB * II));
  size_t r = idx % ((size_t)BB * II);
  int m = (int)(r / II), k = (int)(r % II);
  xbf[idx] = (__bf16)seq[((size_t)m * SS + t) * II + k];  // [b][t][k] -> [t][b][k]
}

__global__ void k_init_state(__bf16* __restrict__ hs0, float* __restrict__ c) {
  size_t idx = (size_t)blockIdx.x * blockDim.x + threadIdx.x;
  if (idx >= N_C) return;
  hs0[idx] = (__bf16)0.0f;
  c[idx] = 0.0f;
}

// ---------------- per-step kernel A: forget-MLP bottleneck ----------------
__global__ void k_mlp_front(int t, const __bf16* __restrict__ xbf, const __bf16* __restrict__ hs_prev,
                            const __bf16* __restrict__ wt1, const float* __restrict__ b1,
                            const float* __restrict__ W2, const float* __restrict__ b2,
                            float* __restrict__ a2g) {
  __shared__ float a1s[BB][M1];
  const int lane = threadIdx.x & 31, wave = threadIdx.x >> 5;
  const int hi = lane >> 4, ln = lane & 15;
  const int mt = wave;                       // 8 waves * 16 rows = 128
  const int mrow = mt * 16 + ln;
  const __bf16* xrow = xbf + ((size_t)t * BB + mrow) * II;
  const __bf16* hrow = hs_prev + (size_t)mrow * HH;
  const __bf16* brow = wt1 + (size_t)ln * KC;
  v8f acc = {};
#pragma unroll
  for (int kt = 0; kt < II / 32; ++kt) {
    v16bf a = load_a16(xrow, kt * 32, hi);
    v16bf b = load_b16(brow + kt * 32 + hi * 16);
    acc = wmma_bf16(a, b, acc);
  }
#pragma unroll 4
  for (int kt = 0; kt < HH / 32; ++kt) {
    v16bf a = load_a16(hrow, kt * 32, hi);
    v16bf b = load_b16(brow + II + kt * 32 + hi * 16);
    acc = wmma_bf16(a, b, acc);
  }
#pragma unroll
  for (int v = 0; v < 8; ++v) {
    int m = mt * 16 + v + 8 * hi;
    a1s[m][ln] = fmaxf(acc[v] + b1[ln], 0.0f);
  }
  __syncthreads();
  for (int idx = threadIdx.x; idx < BB * M2; idx += blockDim.x) {
    int m = idx >> 3, j = idx & 7;
    float s = b2[j];
#pragma unroll
    for (int p = 0; p < M1; ++p) s += a1s[m][p] * W2[p * M2 + j];
    a2g[idx] = fmaxf(s, 0.0f);
  }
}

// ---------------- per-step kernel B: fused i/g/o WMMA + cell update ----------------
// One block per nt (64 blocks); 8 waves = 8 m-tiles. Block's weight slice
// (3 gates x 16 cols x K) is streamed through double-buffered LDS with
// global_load_async_to_lds_b128 (ASYNCcnt), overlapping WMMA on the live buffer.
__global__ void __launch_bounds__(256)
k_lstm_step(int t, const __bf16* __restrict__ xbf, const __bf16* __restrict__ hs_prev,
            const __bf16* __restrict__ wtg, const float* __restrict__ a2g,
            const float* __restrict__ W3, const float* __restrict__ b3,
            const float* __restrict__ b_i, const float* __restrict__ b_c, const float* __restrict__ b_o,
            float* __restrict__ c_state, __bf16* __restrict__ hs_next, float* __restrict__ fs_out) {
  __shared__ __align__(32) __bf16 wbuf[2][WROWS * KCHP];   // 2 x ~25KB, padded rows
  const int tid  = threadIdx.x;
  const int lane = tid & 31, wave = tid >> 5;
  const int nt = blockIdx.x;                 // 64 blocks: one 16-col slice each
  const int mt = wave;                       // 8 m-tiles
  const int hi = lane >> 4, ln = lane & 15;
  const int mrow = mt * 16 + ln;
  const int ncol = nt * 16 + ln;
  const __bf16* xrow = xbf + ((size_t)t * BB + mrow) * II;
  const __bf16* hrow = hs_prev + (size_t)mrow * HH;
  // global weight base for this block: row r (= g*16 + j) -> wtg[(g*HH + nt*16 + j)*KC]
  const __bf16* wgbase = wtg + (size_t)nt * 16 * KC;

  // cooperative async copy of one K chunk into buf: 1536 x 16B, 6 per thread
  auto issue_chunk = [&](int c, int buf) {
#pragma unroll
    for (int u = 0; u < 6; ++u) {
      int unit = tid + u * 256;            // 0..1535
      int r    = unit >> 5;                // 0..47  (32 x 16B per row)
      int off8 = (unit & 31) * 8;          // element offset in row chunk
      uint32_t lds_off = (uint32_t)(uintptr_t)&wbuf[buf][r * KCHP + off8];
      const __bf16* g = wgbase + ((size_t)(r >> 4) * HH + (r & 15)) * KC + (size_t)c * KCH + off8;
      async_g2l_b128(lds_off, g);
    }
  };

  v8f ai = {}, ag = {}, ao = {};
  issue_chunk(0, 0);
#pragma unroll
  for (int c = 0; c < NCHK; ++c) {
    if (c + 1 < NCHK) {
      issue_chunk(c + 1, (c + 1) & 1);
      wait_async_le1();                    // chunk c landed (in-order completion)
    } else {
      wait_async_0();
    }
    __syncthreads();                       // all waves' copies visible
    const __bf16* abase = (c == 0) ? xrow : (hrow + (size_t)(c - 1) * KCH);
    const __bf16* wb = &wbuf[c & 1][0];
    const __bf16* wi = wb + (0 * 16 + ln) * KCHP + hi * 16;
    const __bf16* wc = wb + (1 * 16 + ln) * KCHP + hi * 16;
    const __bf16* wo = wb + (2 * 16 + ln) * KCHP + hi * 16;
#pragma unroll
    for (int ktl = 0; ktl < KCH / 32; ++ktl) {
      const int k0 = ktl * 32;
      v16bf a  = load_a16(abase, k0, hi);
      v16bf bi = load_b16(wi + k0);
      v16bf bc = load_b16(wc + k0);
      v16bf bo = load_b16(wo + k0);
      ai = wmma_bf16(a, bi, ai);
      ag = wmma_bf16(a, bc, ag);
      ao = wmma_bf16(a, bo, ao);
    }
    __syncthreads();                       // safe to overwrite buf[c&1] next round
  }

  // elementwise epilogue: f from tiny MLP, cell/state update
  float w3c[M2];
#pragma unroll
  for (int p = 0; p < M2; ++p) w3c[p] = W3[(size_t)p * HH + ncol];
  const float b3n = b3[ncol], bin = b_i[ncol], bcn = b_c[ncol], bon = b_o[ncol];
#pragma unroll
  for (int v = 0; v < 8; ++v) {
    const int m = mt * 16 + v + 8 * hi;
    float iv = sigmoidf_(ai[v] + bin);
    float gv = tanhf(ag[v] + bcn);
    float ov = sigmoidf_(ao[v] + bon);
    float s = b3n;
    const float* a2r = a2g + (size_t)m * M2;
#pragma unroll
    for (int p = 0; p < M2; ++p) s += a2r[p] * w3c[p];
    float f = sigmoidf_(s);
    const size_t cidx = (size_t)m * HH + ncol;
    float cn = f * c_state[cidx] + iv * gv;
    c_state[cidx] = cn;
    float hn = ov * tanhf(cn);
    hs_next[cidx] = (__bf16)hn;
    fs_out[((size_t)m * SS + t) * HH + ncol] = f;   // fs[b][t][h]
  }
}

// ---------------- final projection: [S*B,1024] @ [1024,256] ----------------
__global__ void __launch_bounds__(256)
k_out_proj(const __bf16* __restrict__ hsbf, const __bf16* __restrict__ woutT,
           const float* __restrict__ b_out, float* __restrict__ pred) {
  const int lane = threadIdx.x & 31, wave = threadIdx.x >> 5;
  const int wg = blockIdx.x * 8 + wave;      // 65536 tiles: 4096 (M) x 16 (N)
  const int ntile = wg & 15;
  const int MT = wg >> 4;
  const int hi = lane >> 4, ln = lane & 15;
  const int mrow = MT * 16 + ln;
  const __bf16* arow = hsbf + (size_t)mrow * HH;
  const __bf16* brow = woutT + (size_t)(ntile * 16 + ln) * HH;
  v8f acc = {};
#pragma unroll 4
  for (int kt = 0; kt < HH / 32; ++kt) {
    v16bf a = load_a16(arow, kt * 32, hi);
    v16bf b = load_b16(brow + kt * 32 + hi * 16);
    acc = wmma_bf16(a, b, acc);
  }
  const float bo = b_out[ntile * 16 + ln];
#pragma unroll
  for (int v = 0; v < 8; ++v) {
    int mr = MT * 16 + v + 8 * hi;
    int tt = mr >> 7, bidx = mr & 127;       // mrow = t*128 + b
    pred[((size_t)bidx * SS + tt) * OO + ntile * 16 + ln] = acc[v] + bo;
  }
}

// ---------------- host launch ----------------
extern "C" void kernel_launch(void* const* d_in, const int* in_sizes, int n_in,
                              void* d_out, int out_size, void* d_ws, size_t ws_size,
                              hipStream_t stream) {
  const float* seq  = (const float*)d_in[0];
  const float* W1   = (const float*)d_in[1];  const float* b1  = (const float*)d_in[2];
  const float* W2   = (const float*)d_in[3];  const float* b2  = (const float*)d_in[4];
  const float* W3   = (const float*)d_in[5];  const float* b3  = (const float*)d_in[6];
  const float* W_hi = (const float*)d_in[7];  const float* W_xi = (const float*)d_in[8];
  const float* b_i  = (const float*)d_in[9];
  const float* W_hc = (const float*)d_in[10]; const float* W_xc = (const float*)d_in[11];
  const float* b_c  = (const float*)d_in[12];
  const float* W_ho = (const float*)d_in[13]; const float* W_xo = (const float*)d_in[14];
  const float* b_o  = (const float*)d_in[15];
  const float* W_out = (const float*)d_in[16]; const float* b_out = (const float*)d_in[17];

  char* ws = (char*)d_ws;
  __bf16* wtg    = (__bf16*)(ws + OFF_WTG);
  __bf16* wt1    = (__bf16*)(ws + OFF_WT1);
  __bf16* woutT  = (__bf16*)(ws + OFF_WOUT);
  __bf16* xbf    = (__bf16*)(ws + OFF_XBF);
  __bf16* hsbase = (__bf16*)(ws + OFF_HS);
  float*  cstate = (float*) (ws + OFF_C);
  float*  a2g    = (float*) (ws + OFF_A2);

  float* pred = (float*)d_out;
  float* fs   = pred + (size_t)BB * SS * OO;

  // one-time packing / init
  k_pack_gate_weights<<<(int)((N_WTG + 255) / 256), 256, 0, stream>>>(W_xi, W_hi, W_xc, W_hc, W_xo, W_ho, wtg);
  k_pack_w1<<<(int)((N_WT1 + 255) / 256), 256, 0, stream>>>(W1, wt1);
  k_pack_wout<<<(int)((N_WOUT + 255) / 256), 256, 0, stream>>>(W_out, woutT);
  k_pack_x<<<(int)((N_XBF + 255) / 256), 256, 0, stream>>>(seq, xbf);
  k_init_state<<<(int)((N_C + 255) / 256), 256, 0, stream>>>(hsbase, cstate);

  const size_t stepN = (size_t)BB * HH;   // 131072 bf16 per h slot
  for (int t = 0; t < SS; ++t) {
    const __bf16* hprev = hsbase + (size_t)t * stepN;
    __bf16* hnext = hsbase + (size_t)(t + 1) * stepN;
    k_mlp_front<<<1, 256, 0, stream>>>(t, xbf, hprev, wt1, b1, W2, b2, a2g);
    k_lstm_step<<<64, 256, 0, stream>>>(t, xbf, hprev, wtg, a2g, W3, b3, b_i, b_c, b_o,
                                        cstate, hnext, fs);
  }
  k_out_proj<<<8192, 256, 0, stream>>>(hsbase + stepN, woutT, b_out, pred);
}
